// DNAStructureAttention_52536039964867
// MI455X (gfx1250) — compile-verified, matrix operand order checked by reading, not run on hardware
//
#include <hip/hip_runtime.h>

// ---------------------------------------------------------------------------
// DNA structure attention for MI455X (gfx1250, wave32, WMMA f16->f32)
//   DIM=1024, HEADS=16, HEAD_DIM=64, B=4, T=2048
// ---------------------------------------------------------------------------

#define DIM      1024
#define HEADS    16
#define HEAD_DIM 64
#define BB       4
#define TT       2048
#define MTOT     (BB * TT)          // 8192 rows
#define QKV_N    (3 * DIM)          // 3072

typedef __attribute__((ext_vector_type(16))) _Float16 v16h;
typedef __attribute__((ext_vector_type(8)))  _Float16 v8h;
typedef __attribute__((ext_vector_type(8)))  float    v8f;
typedef __attribute__((ext_vector_type(4)))  float    v4f;

// Load one WMMA A/B fragment (16x32 f16, K-major row in memory).
// Per ISA 7.12.2: lanes 0-15 hold K = {k0..k0+7, k0+16..k0+23},
// lanes 16-31 hold K = {k0+8..k0+15, k0+24..k0+31}.  `off` = (lane>=16)?8:0.
__device__ __forceinline__ v16h load_frag(const _Float16* __restrict__ rowp,
                                          int k0, int off) {
  v8h lo = *(const v8h*)(rowp + k0 + off);
  v8h hi = *(const v8h*)(rowp + k0 + off + 16);
  return __builtin_shufflevector(lo, hi, 0, 1, 2, 3, 4, 5, 6, 7,
                                         8, 9, 10, 11, 12, 13, 14, 15);
}

__device__ __forceinline__ v8f wmma_f16(v16h a, v16h b, v8f c) {
  return __builtin_amdgcn_wmma_f32_16x16x32_f16(false, a, false, b,
                                                (short)0, c, false, false);
}

// ---------------------------------------------------------------------------
// Kernel 1: vectorized f32 -> f16 conversion (8 elems / thread, b128 traffic)
// ---------------------------------------------------------------------------
__global__ void cvt_f32_to_f16_v8(const float* __restrict__ in,
                                  _Float16* __restrict__ out, int n) {
  const int i = (blockIdx.x * blockDim.x + threadIdx.x) * 8;
  if (i < n) {
    v4f a = *(const v4f*)(in + i);
    v4f b = *(const v4f*)(in + i + 4);
    v8h r;
    r[0] = (_Float16)a[0]; r[1] = (_Float16)a[1];
    r[2] = (_Float16)a[2]; r[3] = (_Float16)a[3];
    r[4] = (_Float16)b[0]; r[5] = (_Float16)b[1];
    r[6] = (_Float16)b[2]; r[7] = (_Float16)b[3];
    *(v8h*)(out + i) = r;
  }
}

// ---------------------------------------------------------------------------
// Kernel 2: QKV projection GEMM  (M=8192, N=3072, K=1024)
//   qkv[m,n] = sum_k x[m,k] * qkv_w[n,k] + qkv_b[n]
// One wave per 32x64 output tile (2 A-frags x 4 B-frags -> 8 WMMA / K-step).
// Writes q -> [B,H,T,d], k -> [B,H,T,d], v -> [B,H,d,T] (transposed).
// ---------------------------------------------------------------------------
__global__ void qkv_gemm(const _Float16* __restrict__ xh,
                         const _Float16* __restrict__ wh,
                         const float* __restrict__ bias,
                         _Float16* __restrict__ qh,
                         _Float16* __restrict__ kh,
                         _Float16* __restrict__ vt) {
  const int lane = threadIdx.x & 31;
  const int wave = threadIdx.x >> 5;
  const int wid  = blockIdx.x * 8 + wave;   // 12288 waves = 256 * 48
  const int mt   = wid & 255;               // M tile (32 rows)
  const int nt   = wid >> 8;                // N tile (0..47)
  const int m0   = mt * 32;
  const int n0   = nt * 64;
  const int lcol = lane & 15;
  const int half = lane >> 4;
  const int off  = half * 8;

  const _Float16* ar0 = xh + (size_t)(m0 + lcol) * DIM;
  const _Float16* ar1 = ar0 + (size_t)16 * DIM;
  const _Float16* br0 = wh + (size_t)(n0 + lcol) * DIM;
  const _Float16* br1 = br0 + (size_t)16 * DIM;
  const _Float16* br2 = br0 + (size_t)32 * DIM;
  const _Float16* br3 = br0 + (size_t)48 * DIM;

  v8f acc[2][4];
#pragma unroll
  for (int mi = 0; mi < 2; ++mi)
#pragma unroll
    for (int s = 0; s < 4; ++s) acc[mi][s] = (v8f){};

  for (int k = 0; k < DIM; k += 32) {
    v16h a0 = load_frag(ar0, k, off);
    v16h a1 = load_frag(ar1, k, off);
    v16h b0 = load_frag(br0, k, off);
    v16h b1 = load_frag(br1, k, off);
    v16h b2 = load_frag(br2, k, off);
    v16h b3 = load_frag(br3, k, off);
    acc[0][0] = wmma_f16(a0, b0, acc[0][0]);
    acc[0][1] = wmma_f16(a0, b1, acc[0][1]);
    acc[0][2] = wmma_f16(a0, b2, acc[0][2]);
    acc[0][3] = wmma_f16(a0, b3, acc[0][3]);
    acc[1][0] = wmma_f16(a1, b0, acc[1][0]);
    acc[1][1] = wmma_f16(a1, b1, acc[1][1]);
    acc[1][2] = wmma_f16(a1, b2, acc[1][2]);
    acc[1][3] = wmma_f16(a1, b3, acc[1][3]);
  }

#pragma unroll
  for (int s = 0; s < 4; ++s) {
    const int n     = n0 + 16 * s + lcol;
    const float bv  = bias[n];
    const int which = n >> 10;        // 0=q 1=k 2=v
    const int rem   = n & 1023;
    const int h     = rem >> 6;
    const int dd    = rem & 63;
#pragma unroll
    for (int mi = 0; mi < 2; ++mi) {
#pragma unroll
      for (int i = 0; i < 8; ++i) {
        const int m = m0 + 16 * mi + i + half * 8;   // C-layout row
        const int b = m >> 11;                       // / T
        const int t = m & 2047;
        const _Float16 hv = (_Float16)(acc[mi][s][i] + bv);
        const size_t bh = (size_t)(b * HEADS + h);
        if (which == 0)      qh[(bh * TT + t) * HEAD_DIM + dd] = hv;
        else if (which == 1) kh[(bh * TT + t) * HEAD_DIM + dd] = hv;
        else                 vt[(bh * HEAD_DIM + dd) * TT + t] = hv;
      }
    }
  }
}

// ---------------------------------------------------------------------------
// Kernel 3: causal flash attention with additive bias.
// One wave per (b,h, 32-query tile); 4 waves / block; online softmax over
// 32-key chunks.  K/V fragments are reused across both 16-query subtiles.
// ---------------------------------------------------------------------------
__global__ void attn_flash(const _Float16* __restrict__ qh,
                           const _Float16* __restrict__ kh,
                           const _Float16* __restrict__ vt,
                           const float* __restrict__ bias2d,
                           _Float16* __restrict__ oh) {
  __shared__ float plds[4][32 * 32];   // per-wave P staging (C -> A layout)

  const int lane = threadIdx.x & 31;
  const int wave = threadIdx.x >> 5;
  const int wid  = blockIdx.x * 4 + wave;  // 4096 waves = 64 (b,h) * 64 qtiles
  const int bh   = wid >> 6;               // (b*H + h)
  const int qt   = wid & 63;
  const int q0   = qt * 32;
  const int b    = bh >> 4;
  const int h    = bh & 15;
  const int lcol = lane & 15;
  const int half = lane >> 4;
  const int off  = half * 8;

  // Q fragments (A-layout): two 16-query subtiles, d split 2x32.
  v16h qa[2][2];
#pragma unroll
  for (int qs = 0; qs < 2; ++qs) {
    const _Float16* qrow =
        qh + ((size_t)bh * TT + (q0 + 16 * qs + lcol)) * HEAD_DIM;
    qa[qs][0] = load_frag(qrow, 0, off);
    qa[qs][1] = load_frag(qrow, 32, off);
  }

  float m_acc[2][8], l_acc[2][8];
#pragma unroll
  for (int qs = 0; qs < 2; ++qs)
#pragma unroll
    for (int i = 0; i < 8; ++i) { m_acc[qs][i] = -1.0e30f; l_acc[qs][i] = 0.0f; }
  v8f o[2][4];
#pragma unroll
  for (int qs = 0; qs < 2; ++qs)
#pragma unroll
    for (int s = 0; s < 4; ++s) o[qs][s] = (v8f){};

  const int nchunk = (q0 >> 5) + 1;    // 32-key chunks covering keys <= q0+31
  for (int c = 0; c < nchunk; ++c) {
    const int kb = c * 32;

    // ---- K fragments (shared by both query subtiles) ----
    const _Float16* krow0 = kh + ((size_t)bh * TT + (kb + lcol)) * HEAD_DIM;
    const _Float16* krow1 = krow0 + (size_t)16 * HEAD_DIM;
    v16h kf[2][2];
    kf[0][0] = load_frag(krow0, 0, off);
    kf[0][1] = load_frag(krow0, 32, off);
    kf[1][0] = load_frag(krow1, 0, off);
    kf[1][1] = load_frag(krow1, 32, off);

    // ---- scores: 2 qsubs x 2 ksubs x (K=64 as 2x32) = 8 WMMAs ----
    v8f s[2][2];
#pragma unroll
    for (int qs = 0; qs < 2; ++qs) {
#pragma unroll
      for (int ks = 0; ks < 2; ++ks) {
        v8f t = {};
        t = wmma_f16(qa[qs][0], kf[ks][0], t);
        t = wmma_f16(qa[qs][1], kf[ks][1], t);
        s[qs][ks] = t;
      }
    }

    // ---- per-subtile: scale + bias + mask + online softmax update ----
    float* pb = plds[wave];
#pragma unroll
    for (int qs = 0; qs < 2; ++qs) {
      float mx[8];
#pragma unroll
      for (int i = 0; i < 8; ++i) {
        const int qr  = q0 + 16 * qs + i + half * 8;
        const int k0i = kb + lcol;
        const int k1i = kb + 16 + lcol;
        float v0 = s[qs][0][i] * 0.125f + bias2d[(size_t)qr * TT + k0i];
        float v1 = s[qs][1][i] * 0.125f + bias2d[(size_t)qr * TT + k1i];
        if (k0i > qr) v0 = -3.0e38f;
        if (k1i > qr) v1 = -3.0e38f;
        s[qs][0][i] = v0; s[qs][1][i] = v1;
        mx[i] = fmaxf(v0, v1);
      }
#pragma unroll
      for (int d = 1; d < 16; d <<= 1) {
#pragma unroll
        for (int i = 0; i < 8; ++i)
          mx[i] = fmaxf(mx[i], __shfl_xor(mx[i], d));
      }
      float al[8], rs[8];
#pragma unroll
      for (int i = 0; i < 8; ++i) {
        const float nm = fmaxf(m_acc[qs][i], mx[i]);
        al[i] = __expf(m_acc[qs][i] - nm);
        m_acc[qs][i] = nm;
        s[qs][0][i] = __expf(s[qs][0][i] - nm);
        s[qs][1][i] = __expf(s[qs][1][i] - nm);
        rs[i] = s[qs][0][i] + s[qs][1][i];
      }
#pragma unroll
      for (int d = 1; d < 16; d <<= 1) {
#pragma unroll
        for (int i = 0; i < 8; ++i)
          rs[i] += __shfl_xor(rs[i], d);
      }
#pragma unroll
      for (int i = 0; i < 8; ++i) {
        l_acc[qs][i] = l_acc[qs][i] * al[i] + rs[i];
        o[qs][0][i] *= al[i]; o[qs][1][i] *= al[i];
        o[qs][2][i] *= al[i]; o[qs][3][i] *= al[i];
      }
      // P (C-layout) -> LDS rows 16*qs .. 16*qs+15
#pragma unroll
      for (int i = 0; i < 8; ++i) {
        const int r = 16 * qs + i + half * 8;
        pb[r * 32 + lcol]      = s[qs][0][i];
        pb[r * 32 + 16 + lcol] = s[qs][1][i];
      }
    }
    asm volatile("s_wait_dscnt 0x0" ::: "memory");

    // ---- read P back as A-layout f16 fragments ----
    v16h pa[2];
#pragma unroll
    for (int qs = 0; qs < 2; ++qs) {
      const float* prow = pb + (16 * qs + lcol) * 32;
#pragma unroll
      for (int j = 0; j < 8; ++j) {
        pa[qs][j]     = (_Float16)prow[off + j];
        pa[qs][8 + j] = (_Float16)prow[16 + off + j];
      }
    }

    // ---- V fragments (shared), O += P @ V : 8 WMMAs ----
    const _Float16* vbase = vt + (size_t)bh * HEAD_DIM * TT + kb;
    v16h vb[4];
    vb[0] = load_frag(vbase + (size_t)(0  + lcol) * TT, 0, off);
    vb[1] = load_frag(vbase + (size_t)(16 + lcol) * TT, 0, off);
    vb[2] = load_frag(vbase + (size_t)(32 + lcol) * TT, 0, off);
    vb[3] = load_frag(vbase + (size_t)(48 + lcol) * TT, 0, off);
#pragma unroll
    for (int qs = 0; qs < 2; ++qs) {
#pragma unroll
      for (int sd = 0; sd < 4; ++sd)
        o[qs][sd] = wmma_f16(pa[qs], vb[sd], o[qs][sd]);
    }
  }

  // ---- epilogue: O / l, f16 store into [B,T,DIM] with head interleave ----
#pragma unroll
  for (int qs = 0; qs < 2; ++qs) {
#pragma unroll
    for (int sd = 0; sd < 4; ++sd) {
      const int colC = h * HEAD_DIM + 16 * sd + lcol;
#pragma unroll
      for (int i = 0; i < 8; ++i) {
        const int m = b * TT + q0 + 16 * qs + i + half * 8;
        oh[(size_t)m * DIM + colC] = (_Float16)(o[qs][sd][i] / l_acc[qs][i]);
      }
    }
  }
}

// ---------------------------------------------------------------------------
// Kernel 4: output projection GEMM (M=8192, N=1024, K=1024), f32 out + bias.
// One wave per 32x64 output tile.
// ---------------------------------------------------------------------------
__global__ void out_gemm(const _Float16* __restrict__ ah,
                         const _Float16* __restrict__ wh,
                         const float* __restrict__ bias,
                         float* __restrict__ out) {
  const int lane = threadIdx.x & 31;
  const int wave = threadIdx.x >> 5;
  const int wid  = blockIdx.x * 8 + wave;   // 4096 waves = 256 * 16
  const int mt   = wid & 255;
  const int nt   = wid >> 8;
  const int m0   = mt * 32;
  const int n0   = nt * 64;
  const int lcol = lane & 15;
  const int half = lane >> 4;
  const int off  = half * 8;

  const _Float16* ar0 = ah + (size_t)(m0 + lcol) * DIM;
  const _Float16* ar1 = ar0 + (size_t)16 * DIM;
  const _Float16* br0 = wh + (size_t)(n0 + lcol) * DIM;
  const _Float16* br1 = br0 + (size_t)16 * DIM;
  const _Float16* br2 = br0 + (size_t)32 * DIM;
  const _Float16* br3 = br0 + (size_t)48 * DIM;

  v8f acc[2][4];
#pragma unroll
  for (int mi = 0; mi < 2; ++mi)
#pragma unroll
    for (int s = 0; s < 4; ++s) acc[mi][s] = (v8f){};

  for (int k = 0; k < DIM; k += 32) {
    v16h a0 = load_frag(ar0, k, off);
    v16h a1 = load_frag(ar1, k, off);
    v16h b0 = load_frag(br0, k, off);
    v16h b1 = load_frag(br1, k, off);
    v16h b2 = load_frag(br2, k, off);
    v16h b3 = load_frag(br3, k, off);
    acc[0][0] = wmma_f16(a0, b0, acc[0][0]);
    acc[0][1] = wmma_f16(a0, b1, acc[0][1]);
    acc[0][2] = wmma_f16(a0, b2, acc[0][2]);
    acc[0][3] = wmma_f16(a0, b3, acc[0][3]);
    acc[1][0] = wmma_f16(a1, b0, acc[1][0]);
    acc[1][1] = wmma_f16(a1, b1, acc[1][1]);
    acc[1][2] = wmma_f16(a1, b2, acc[1][2]);
    acc[1][3] = wmma_f16(a1, b3, acc[1][3]);
  }

#pragma unroll
  for (int s = 0; s < 4; ++s) {
    const int n = n0 + 16 * s + lcol;
    const float bv = bias[n];
#pragma unroll
    for (int mi = 0; mi < 2; ++mi) {
#pragma unroll
      for (int i = 0; i < 8; ++i) {
        const int m = m0 + 16 * mi + i + half * 8;
        out[(size_t)m * DIM + n] = acc[mi][s][i] + bv;
      }
    }
  }
}

// ---------------------------------------------------------------------------
// Host launcher
// ---------------------------------------------------------------------------
extern "C" void kernel_launch(void* const* d_in, const int* in_sizes, int n_in,
                              void* d_out, int out_size, void* d_ws, size_t ws_size,
                              hipStream_t stream) {
  (void)in_sizes; (void)n_in; (void)out_size; (void)ws_size;

  const float* x        = (const float*)d_in[0];   // [B,T,DIM]
  const float* qkv_w    = (const float*)d_in[1];   // [3*DIM, DIM]
  const float* qkv_b    = (const float*)d_in[2];   // [3*DIM]
  const float* out_w    = (const float*)d_in[3];   // [DIM, DIM]
  const float* out_b    = (const float*)d_in[4];   // [DIM]
  const float* dna_bias = (const float*)d_in[5];   // [2048, 2048]
  float* out = (float*)d_out;                      // [B,T,DIM] f32

  char* ws = (char*)d_ws;
  // Workspace map (bytes):
  _Float16* xh     = (_Float16*)(ws + 0);           // 16,777,216
  _Float16* qkvwh  = (_Float16*)(ws + 16777216);    //  6,291,456
  _Float16* outwh  = (_Float16*)(ws + 23068672);    //  2,097,152
  _Float16* qh     = (_Float16*)(ws + 25165824);    // 16,777,216
  _Float16* kh     = (_Float16*)(ws + 41943040);    // 16,777,216
  _Float16* vt     = (_Float16*)(ws + 58720256);    // 16,777,216
  _Float16* attnh  = (_Float16*)(ws + 75497472);    // 16,777,216
  // total: 92,274,688 bytes

  const int nx = MTOT * DIM;        // 8,388,608
  const int nw = QKV_N * DIM;       // 3,145,728
  const int no = DIM * DIM;         // 1,048,576

  cvt_f32_to_f16_v8<<<nx / 8 / 256, 256, 0, stream>>>(x, xh, nx);
  cvt_f32_to_f16_v8<<<nw / 8 / 256, 256, 0, stream>>>(qkv_w, qkvwh, nw);
  cvt_f32_to_f16_v8<<<no / 8 / 256, 256, 0, stream>>>(out_w, outwh, no);

  // 12288 wave-tiles (32x64 each), 8 waves (256 threads) per block
  qkv_gemm<<<1536, 256, 0, stream>>>(xh, qkvwh, qkv_b, qh, kh, vt);

  // 4096 wave-tiles (32 queries each), 4 waves (128 threads) per block
  attn_flash<<<1024, 128, 0, stream>>>(qh, kh, vt, dna_bias, attnh);

  // 4096 wave-tiles (32x64 each), 8 waves per block
  out_gemm<<<512, 256, 0, stream>>>(attnh, outwh, out_b, out);
}
